// TaskAlignedAssigner_10763188044393
// MI455X (gfx1250) — compile-verified, matrix operand order checked by reading, not run on hardware
//
#include <hip/hip_runtime.h>
#include <hip/hip_bf16.h>

typedef __attribute__((ext_vector_type(8))) int v8i;

#define EPS9 1e-9f
#define EPS7 1e-7f
#define TOPK_K 13
#define INV_PI2 0.40528473456935109f /* 4/pi^2 */

// ---------------------------------------------------------------- small init
__global__ void k_init_pos(float* pos_am, float* pos_ov, int n) {
  int i = blockIdx.x * blockDim.x + threadIdx.x;
  if (i < n) { pos_am[i] = 0.f; pos_ov[i] = 0.f; }
}

__global__ void k_zero_words(unsigned int* __restrict__ p, long long nwords) {
  long long i = (long long)blockIdx.x * blockDim.x + threadIdx.x;
  if (i < nwords) p[i] = 0u;
}

// ------------------------------------------- precompute atan(w2/h2) per (b,a)
__global__ void k_at2(const float* __restrict__ pd_bboxes, float* __restrict__ at2, int total) {
  int i = blockIdx.x * blockDim.x + threadIdx.x;
  if (i >= total) return;
  const float* p = pd_bboxes + (size_t)i * 4;
  float w2 = p[2] - p[0];
  float h2 = p[3] - p[1] + EPS7;
  at2[i] = atanf(w2 / h2);
}

// -------------------- per (b,m,a): mask, CIoU -> overlaps, align_metric store
__global__ void k_pair(const float* __restrict__ pd_scores, const float* __restrict__ pd_bboxes,
                       const float* __restrict__ anc, const int* __restrict__ gt_labels,
                       const float* __restrict__ gt_bboxes, const float* __restrict__ mask_gt,
                       const float* __restrict__ at2, float* __restrict__ overlaps,
                       float* __restrict__ align_m, int M, int na, int nc) {
  int bm = blockIdx.x;            // b*M + m
  int b  = bm / M;
  const float* g = gt_bboxes + (size_t)bm * 4;
  float gx1 = g[0], gy1 = g[1], gx2 = g[2], gy2 = g[3];
  float mg  = mask_gt[bm];
  int   lbl = gt_labels[bm];
  float w1  = gx2 - gx1, h1 = gy2 - gy1 + EPS7;
  float at1 = atanf(w1 / h1);
  float a1area = w1 * h1;
  size_t rowo = (size_t)bm * na;
  size_t bo   = (size_t)b  * na;
  for (int a = threadIdx.x; a < na; a += blockDim.x) {
    float ax = anc[a * 2 + 0], ay = anc[a * 2 + 1];
    float dmin = fminf(fminf(ax - gx1, ay - gy1), fminf(gx2 - ax, gy2 - ay));
    bool mask = (dmin > EPS9) && (mg > 0.f);
    const float* p = pd_bboxes + (bo + a) * 4;
    float px1 = p[0], py1 = p[1], px2 = p[2], py2 = p[3];
    float w2 = px2 - px1, h2 = py2 - py1 + EPS7;
    float iw = fminf(gx2, px2) - fmaxf(gx1, px1);
    float ih = fminf(gy2, py2) - fmaxf(gy1, py1);
    float inter = fmaxf(iw, 0.f) * fmaxf(ih, 0.f);
    float uni   = a1area + w2 * h2 - inter + EPS7;
    float iou   = inter / uni;
    float cw = fmaxf(gx2, px2) - fminf(gx1, px1);
    float ch = fmaxf(gy2, py2) - fminf(gy1, py1);
    float c2 = cw * cw + ch * ch + EPS7;
    float dx = px1 + px2 - gx1 - gx2;
    float dy = py1 + py2 - gy1 - gy2;
    float rho2 = (dx * dx + dy * dy) * 0.25f;
    float dv = at2[bo + a] - at1;
    float v  = INV_PI2 * dv * dv;
    float alpha = v / (v - iou + 1.0f + EPS7);
    float ciou  = iou - (rho2 / c2 + v * alpha);
    float ov = mask ? fmaxf(ciou, 0.f) : 0.f;
    float sc = mask ? pd_scores[(bo + a) * (size_t)nc + lbl] : 0.f;
    float ov2 = ov * ov;
    overlaps[rowo + a] = ov;
    align_m[rowo + a]  = sc * ov2 * ov2 * ov2;   // alpha=1, beta=6
  }
}

// -------------------------- top-13 per (b,m) row (LDS row + iterative argmax)
// mask_pos layout: [(b*na + a)*M + m]  (K-contiguous for the WMMA consumer)
__global__ void k_topk(const float* __restrict__ align_m, const float* __restrict__ anc,
                       const float* __restrict__ gt_bboxes, const float* __restrict__ mask_gt,
                       unsigned char* __restrict__ mask_pos, int M, int na) {
  extern __shared__ float srow[];
  __shared__ float sval[256];
  __shared__ int   sidx[256];
  int bm  = blockIdx.x;
  int tid = threadIdx.x;
  if (mask_gt[bm] <= 0.f) return;            // uniform (buffer pre-zeroed; ref masks these rows out)
  int b = bm / M;
  int m = bm - b * M;
  const float* g = gt_bboxes + (size_t)bm * 4;
  float gx1 = g[0], gy1 = g[1], gx2 = g[2], gy2 = g[3];
  const float* arow = align_m + (size_t)bm * na;
  for (int a = tid; a < na; a += blockDim.x) srow[a] = arow[a];
  __syncthreads();
  for (int p = 0; p < TOPK_K; ++p) {
    float bv = -1e30f; int bi = 0x7fffffff;
    for (int a = tid; a < na; a += blockDim.x) {
      float v = srow[a];
      if (v > bv || (v == bv && a < bi)) { bv = v; bi = a; }
    }
    sval[tid] = bv; sidx[tid] = bi;
    __syncthreads();
    for (int s = blockDim.x >> 1; s > 0; s >>= 1) {
      if (tid < s) {
        float ov = sval[tid + s]; int oi = sidx[tid + s];
        if (ov > sval[tid] || (ov == sval[tid] && oi < sidx[tid])) { sval[tid] = ov; sidx[tid] = oi; }
      }
      __syncthreads();
    }
    if (tid == 0) {
      int k = sidx[0];
      srow[k] = -2.0f;                                   // remove from candidates
      float ax = anc[k * 2], ay = anc[k * 2 + 1];
      float dmin = fminf(fminf(ax - gx1, ay - gy1), fminf(gx2 - ax, gy2 - ay));
      mask_pos[((size_t)b * na + k) * M + m] =
          (dmin > EPS9) ? (unsigned char)1 : (unsigned char)0;       // * mask_in_gts
    }
    __syncthreads();
  }
}

// fg_sum[b][a] = sum_m mask_pos via integer WMMA:
//   D(16x16 i32) += ones(16x64 u8) x mask(64x16 u8) ; exact, K=64 per op.
// K is memory-contiguous -> B fragment = two aligned b128 loads per lane.
__global__ void k_fgsum_wmma(const unsigned char* __restrict__ mask_pos,
                             float* __restrict__ fg_sum, int bs, int M, int na) {
  int gwave = (blockIdx.x * blockDim.x + threadIdx.x) >> 5;
  int lane  = threadIdx.x & 31;
  int tilesPerB = (na + 15) >> 4;
  int b  = gwave / tilesPerB;
  if (b >= bs) return;                          // uniform per wave
  int a0 = (gwave % tilesPerB) << 4;
  int n  = lane & 15;                           // column N for B operand
  int kb = (lane >> 4) << 4;                    // lanes 0-15: K base 0 ; lanes 16-31: K base 16
  int acol = a0 + n; if (acol > na - 1) acol = na - 1;
  v8i av;
#pragma unroll
  for (int e = 0; e < 8; ++e) av[e] = 0x01010101; // A = all-ones u8 -> D[i][n] = sum_k B[k][n]
  v8i c = {};
  const unsigned char* col = mask_pos + ((size_t)b * na + acol) * M;
  if ((M & 15) == 0) {
    const unsigned int* colw = (const unsigned int*)col;   // 16B-aligned groups
    int kk = 0;
    for (; kk + 64 <= M; kk += 64) {            // full tiles: no guards, vector loads
      int kw = (kk + kb) >> 2;
      v8i bv;
#pragma unroll
      for (int v = 0; v < 4; ++v) bv[v]     = (int)colw[kw + v];      // K = kk+kb   .. +15
#pragma unroll
      for (int v = 0; v < 4; ++v) bv[4 + v] = (int)colw[kw + 8 + v];  // K = kk+kb+32.. +47
      c = __builtin_amdgcn_wmma_i32_16x16x64_iu8(false, av, false, bv, c, false, false);
    }
    if (kk < M) {                                // tail tile (never runs for M=128)
      int kw = (kk + kb) >> 2;
      v8i bv;
#pragma unroll
      for (int v = 0; v < 4; ++v) bv[v]     = (kk + kb + 4 * v      < M) ? (int)colw[kw + v]     : 0;
#pragma unroll
      for (int v = 0; v < 4; ++v) bv[4 + v] = (kk + kb + 4 * v + 32 < M) ? (int)colw[kw + 8 + v] : 0;
      c = __builtin_amdgcn_wmma_i32_16x16x64_iu8(false, av, false, bv, c, false, false);
    }
  } else {                                       // generic fallback (byte path)
    for (int kk = 0; kk < M; kk += 64) {
      v8i bv;
#pragma unroll
      for (int v = 0; v < 8; ++v) {
        int base = kk + kb + ((v & 3) << 2) + ((v >> 2) << 5);
        unsigned int wword = 0;
#pragma unroll
        for (int j = 0; j < 4; ++j) {
          int mm = base + j;
          unsigned int byt = (mm < M) ? (unsigned int)col[mm] : 0u;
          wword |= byt << (8 * j);
        }
        bv[v] = (int)wword;
      }
      c = __builtin_amdgcn_wmma_i32_16x16x64_iu8(false, av, false, bv, c, false, false);
    }
  }
  if (lane < 16 && (a0 + lane) < na)
    fg_sum[(size_t)b * na + a0 + lane] = (float)c[0];   // row M=0, N=lane
}

// ---------- resolve multi-assignment, emit target_gt_idx / final fg, pos maxes
__global__ void k_resolve(const float* __restrict__ overlaps, const float* __restrict__ align_m,
                          const unsigned char* __restrict__ mask_pos,
                          float* __restrict__ fg_sum, int* __restrict__ tgt_idx,
                          float* __restrict__ pos_am, float* __restrict__ pos_ov,
                          int M, int na) {
  int a = blockIdx.x * blockDim.x + threadIdx.x;
  int b = blockIdx.y;
  if (a >= na) return;
  size_t pa = (size_t)b * na + a;
  float fg = fg_sum[pa];
  int tgt = 0; float fgf = 0.f;
  if (fg > 0.5f && fg < 1.5f) {                  // single: first m with mask==1 (contiguous scan)
    fgf = 1.f;
    const unsigned char* col = mask_pos + pa * M;
    for (int m = 0; m < M; ++m) { if (col[m]) { tgt = m; break; } }
  } else if (fg >= 1.5f) {                       // multi: argmax_m overlaps (first max)
    fgf = 1.f;
    const float* col = overlaps + (size_t)b * M * na + a;
    float best = -1.f;
    for (int m = 0; m < M; ++m) {
      float v = col[(size_t)m * na];
      if (v > best) { best = v; tgt = m; }
    }
  }
  tgt_idx[pa] = tgt;
  fg_sum[pa]  = fgf;
  if (fgf > 0.f) {
    size_t e = ((size_t)b * M + tgt) * na + a;
    atomicMax((int*)&pos_am[b * M + tgt], __float_as_int(align_m[e]));  // values >= 0
    atomicMax((int*)&pos_ov[b * M + tgt], __float_as_int(overlaps[e]));
  }
}

// ------------------------------------------------------------- final outputs
__global__ void k_final(const float* __restrict__ align_m, const int* __restrict__ gt_labels,
                        const float* __restrict__ gt_bboxes, const float* __restrict__ fg_sum,
                        const int* __restrict__ tgt_idx, const float* __restrict__ pos_am,
                        const float* __restrict__ pos_ov, float* __restrict__ out,
                        int M, int na, int nc) {
  __shared__ int   s_lbl[256];
  __shared__ float s_nrm[256];
  int tid = threadIdx.x;
  int a0  = blockIdx.x * blockDim.x;
  int a   = a0 + tid;
  int b   = blockIdx.y;
  size_t L = (size_t)gridDim.y * na;       // bs*na
  int n_valid = na - a0; if (n_valid > (int)blockDim.x) n_valid = blockDim.x;

  if (a < na) {
    size_t pa = (size_t)b * na + a;
    int   tgt = tgt_idx[pa];
    float fgf = fg_sum[pa];
    int   lblr = gt_labels[b * M + tgt];
    int   lbl  = lblr > 0 ? lblr : 0;
    float norm = 0.f;
    if (fgf > 0.f) {
      size_t e = ((size_t)b * M + tgt) * na + a;
      norm = align_m[e] * pos_ov[b * M + tgt] / (pos_am[b * M + tgt] + EPS9);
    }
    s_lbl[tid] = lbl;
    s_nrm[tid] = (fgf > 0.f) ? norm : 0.f;
    // 1) target_labels
    out[pa] = (float)lbl;
    // 2) target_bboxes (16B-aligned vector store: d_out 256-aligned, L*4 % 16 == 0)
    const float* g = gt_bboxes + ((size_t)b * M + tgt) * 4;
    float4* ob = (float4*)(out + L + pa * 4);
    *ob = make_float4(g[0], g[1], g[2], g[3]);
    // 4) fg_mask (bool -> float)
    out[85 * L + pa] = fgf;
    // 5) target_gt_idx
    out[86 * L + pa] = (float)tgt;
  }
  __syncthreads();
  // 3) target_scores: contiguous region for this block, fully coalesced writes
  float* os = out + 5 * L + ((size_t)b * na + a0) * (size_t)nc;
  int total = n_valid * nc;
  for (int i = tid; i < total; i += blockDim.x) {
    int al = i / nc;
    int cc = i - al * nc;
    float nv = s_nrm[al];
    os[i] = (cc == s_lbl[al] && nv != 0.f) ? nv : 0.f;
  }
}

static inline size_t rup256(size_t x) { return (x + 255) & ~(size_t)255; }

extern "C" void kernel_launch(void* const* d_in, const int* in_sizes, int n_in,
                              void* d_out, int out_size, void* d_ws, size_t ws_size,
                              hipStream_t stream) {
  const float* pd_scores = (const float*)d_in[0];
  const float* pd_bboxes = (const float*)d_in[1];
  const float* anc       = (const float*)d_in[2];
  const int*   gt_labels = (const int*)  d_in[3];
  const float* gt_bboxes = (const float*)d_in[4];
  const float* mask_gt   = (const float*)d_in[5];
  float* out = (float*)d_out;

  int na = in_sizes[2] / 2;
  int bs = in_sizes[1] / (na * 4);
  int nc = in_sizes[0] / (bs * na);
  int M  = in_sizes[3] / bs;

  size_t NBM = (size_t)bs * M * na;
  size_t L   = (size_t)bs * na;

  char* w = (char*)d_ws;
  float*         overlaps = (float*)w;          w += rup256(NBM * 4);
  float*         align_m  = (float*)w;          w += rup256(NBM * 4);
  unsigned char* mask_pos = (unsigned char*)w;  w += rup256(NBM);
  float*         fg_sum   = (float*)w;          w += rup256(L * 4);
  int*           tgt_idx  = (int*)w;            w += rup256(L * 4);
  float*         pos_am   = (float*)w;          w += rup256((size_t)bs * M * 4);
  float*         pos_ov   = (float*)w;          w += rup256((size_t)bs * M * 4);
  float*         at2      = (float*)w;          w += rup256(L * 4);
  (void)ws_size; (void)n_in; (void)out_size;

  // 0) init pos_am / pos_ov, zero mask_pos (dword-coalesced; allocation is 256B-rounded)
  k_init_pos<<<(bs * M + 255) / 256, 256, 0, stream>>>(pos_am, pos_ov, bs * M);
  {
    long long nwords = (long long)((NBM + 3) / 4);
    k_zero_words<<<(unsigned)((nwords + 255) / 256), 256, 0, stream>>>(
        (unsigned int*)mask_pos, nwords);
  }
  // 1) atan(w2/h2) per (b,a)
  k_at2<<<((int)L + 255) / 256, 256, 0, stream>>>(pd_bboxes, at2, (int)L);
  // 2) pairwise overlaps + align_metric
  k_pair<<<bs * M, 256, 0, stream>>>(pd_scores, pd_bboxes, anc, gt_labels, gt_bboxes,
                                     mask_gt, at2, overlaps, align_m, M, na, nc);
  // 3) top-13 per gt row -> mask_pos (scatters <=13 bytes per row)
  k_topk<<<bs * M, 256, (size_t)na * sizeof(float), stream>>>(align_m, anc, gt_bboxes,
                                                              mask_gt, mask_pos, M, na);
  // 4) fg counts via integer WMMA (u8 masks, i32 accum, exact)
  {
    int tilesPerB = (na + 15) >> 4;
    int waves  = bs * tilesPerB;
    int blocks = (waves + 3) / 4;                 // 4 waves / 128-thread block
    k_fgsum_wmma<<<blocks, 128, 0, stream>>>(mask_pos, fg_sum, bs, M, na);
  }
  // 5) resolve multi-assignment + per-gt maxima
  {
    dim3 g((na + 255) / 256, bs);
    k_resolve<<<g, 256, 0, stream>>>(overlaps, align_m, mask_pos, fg_sum, tgt_idx,
                                     pos_am, pos_ov, M, na);
  }
  // 6) outputs
  {
    dim3 g((na + 255) / 256, bs);
    k_final<<<g, 256, 0, stream>>>(align_m, gt_labels, gt_bboxes, fg_sum, tgt_idx,
                                   pos_am, pos_ov, out, M, na, nc);
  }
}
